// FFTEmbedding_25933012533404
// MI455X (gfx1250) — matrix operand order repaired and checked by reading, not compiled, not run on hardware
//
#include <hip/hip_runtime.h>
#include <hip/hip_bf16.h>

// FFT-embedding == causal sliding-window correlation out = conv(x, M) + bias,
// where M = weight @ rfft_basis (512x256). Computed as an f16 WMMA GEMM:
//   A = windows (131072 x 256, built from an LDS-staged x span)
//   B = M^T     (256 x 512, f16, staged via LDS)
//   C = f32 accum initialized with bias, streamed out with NT stores.

typedef __attribute__((ext_vector_type(16))) _Float16 v16h;
typedef __attribute__((ext_vector_type(8)))  float    v8f;

#ifndef M_PI_F
#define M_PI_F 3.14159265358979323846f
#endif

// ---------------------------------------------------------------------------
// Kernel A: fold the rfft basis into the weight: Mh[e][w] (f16), 512x256.
//   spec[k] = sum_w x[w] e^{-i 2pi k w/256}; feats = [Re(0..128), Im(0..128)]
//   M[e][w] = sum_k wr[k] cos(th) - wi[k] sin(th),  th = 2pi k w / 256
// Using c = -2pi w/256:  cos(c k) = cos(th),  sin(c k) = -sin(th)
//   => M[e][w] = sum_k wr[k] cos(c k) + wi[k] sin(c k)
// ---------------------------------------------------------------------------
__global__ __launch_bounds__(256) void fftemb_build_M(
    const float* __restrict__ weight,   // (512, 258)
    _Float16* __restrict__ Mh)          // (512, 256) f16
{
  __shared__ float wr[129];
  __shared__ float wi[129];
  const int e = blockIdx.x;
  for (int i = threadIdx.x; i < 129; i += 256) {
    wr[i] = weight[e * 258 + i];
    wi[i] = weight[e * 258 + 129 + i];
  }
  __syncthreads();

  const int w = threadIdx.x;                      // 0..255
  const float c = -2.0f * M_PI_F * (float)w / 256.0f;
  float s = wr[0];                                // k = 0 term
  for (int k = 1; k < 129; ++k) {
    float sn, cs;
    __sincosf(c * (float)k, &sn, &cs);
    s += wr[k] * cs + wi[k] * sn;
  }
  Mh[e * 256 + w] = (_Float16)s;
}

// ---------------------------------------------------------------------------
// Kernel B: tiled WMMA GEMM over sliding windows.
//   Block tile: 128 rows (time) x 128 cols (emb). 8 waves, each wave owns
//   16 rows x 128 cols = 8 accumulator fragments (v8f each).
//   K = 256, processed as 2 LDS-staged halves x 4 kk-steps of K=32.
// ---------------------------------------------------------------------------
#define T_LEN   8192
#define EMB     512
#define WSZ     256
#define XSPAN   384          // 255 + 128 (+1 pad)

__global__ __launch_bounds__(256) void fftemb_conv_wmma(
    const float*    __restrict__ x,     // (16, 8192)
    const _Float16* __restrict__ Mh,    // (512, 256)
    const float*    __restrict__ bias,  // (512)
    float*          __restrict__ out)   // (16, 8192, 512)
{
  __shared__ _Float16 sB[128 * 128];    // 32 KB: [eLocal][k within half]
  __shared__ float    sX[XSPAN];        // 1.5 KB: x span (f32)

  const int tid     = threadIdx.x;
  const int cbBlk   = blockIdx.x & 3;          // 4 column blocks of 128
  const int rowTile = blockIdx.x >> 2;         // 1024 row tiles of 128
  const int e0      = cbBlk * 128;
  const int t0g     = rowTile * 128;           // flat row in [0, B*T)
  const int b       = t0g >> 13;               // / 8192
  const int t0      = t0g & (T_LEN - 1);

  // ---- stage x span: sX[i] = x[b, t0 - 255 + i] (0 outside) ----
  for (int i = tid; i < XSPAN; i += 256) {
    const int tt = t0 - 255 + i;
    float v = 0.0f;
    if (tt >= 0 && tt < T_LEN) v = x[b * T_LEN + tt];
    sX[i] = v;
  }

  const int wave = tid >> 5;
  const int lane = tid & 31;
  const int g    = lane >> 4;     // lane-group (K-half select in frags)
  const int rl   = lane & 15;     // A row / B,D column within 16

  // ---- accumulators initialized with bias (C layout: col N = lane%16) ----
  v8f acc[8];
#pragma unroll
  for (int cb = 0; cb < 8; ++cb) {
    const float bv = bias[e0 + cb * 16 + rl];
    v8f a;
#pragma unroll
    for (int v = 0; v < 8; ++v) a[v] = bv;
    acc[cb] = a;
  }

  const int rowA = wave * 16 + rl;   // A-matrix row within tile (both halves)

  for (int half = 0; half < 2; ++half) {
    __syncthreads();   // previous half's sB reads done / sX ready

    // ---- stage M tile half: 128 rows x 128 f16 = 2048 uint4 ----
    {
      const int kbase = half * 128;
      for (int i = tid; i < 2048; i += 256) {
        const int row = i >> 4;          // 16 uint4 per 128-f16 row
        const int col = i & 15;
        ((uint4*)sB)[i] =
            *(const uint4*)(Mh + (e0 + row) * WSZ + kbase + col * 8);
      }
    }
    __syncthreads();

#pragma unroll
    for (int kk2 = 0; kk2 < 4; ++kk2) {
      const int kk = half * 4 + kk2;     // global K step (of 8)

      // ---- build A fragment from LDS x span (16x32 f16 layout) ----
      // per-lane K indices: 16g + j, +32*kk
      v16h afrag;
#pragma unroll
      for (int j = 0; j < 16; ++j) {
        const int k = (16 * g + j) + 32 * kk;
        afrag[j] = (_Float16)sX[rowA + k];
      }

      // ---- 8 column fragments: B per-lane = contiguous K run of 16 ----
#pragma unroll
      for (int cb = 0; cb < 8; ++cb) {
        const int eL = cb * 16 + rl;
        const v16h bfrag =
            *(const v16h*)(sB + eL * 128 + kk2 * 32 + g * 16);  // 32B aligned
        acc[cb] = __builtin_amdgcn_wmma_f32_16x16x32_f16(
            /*neg_a=*/false, afrag, /*neg_b=*/false, bfrag,
            /*c_mod=*/(short)0, acc[cb], /*reuse_a=*/false, /*reuse_b=*/false);
      }
    }
  }

  // ---- store C: lane L, VGPR v -> row = v + 8*(L/16), col = L%16 ----
  // Output (268 MB) is write-once and larger than L2 (192 MB): use
  // non-temporal stores so the streamed C doesn't evict the reused Mh/x.
#pragma unroll
  for (int cb = 0; cb < 8; ++cb) {
#pragma unroll
    for (int v = 0; v < 8; ++v) {
      const int row = wave * 16 + v + 8 * g;
      const long o  = (long)(b * T_LEN + t0 + row) * EMB + (e0 + cb * 16 + rl);
      __builtin_nontemporal_store(acc[cb][v], &out[o]);
    }
  }
}

extern "C" void kernel_launch(void* const* d_in, const int* in_sizes, int n_in,
                              void* d_out, int out_size, void* d_ws, size_t ws_size,
                              hipStream_t stream) {
  const float* x      = (const float*)d_in[0];   // (16, 8192)
  const float* weight = (const float*)d_in[1];   // (512, 258)
  const float* bias   = (const float*)d_in[2];   // (512,)
  float* out          = (float*)d_out;           // (16, 8192, 512)

  _Float16* Mh = (_Float16*)d_ws;                // 512*256*2 = 256 KB scratch

  // Fold DFT basis into weights (cheap; rerun every call -> deterministic).
  fftemb_build_M<<<512, 256, 0, stream>>>(weight, Mh);

  // Main sliding-window WMMA GEMM: 4096 blocks x 256 threads.
  fftemb_conv_wmma<<<4096, 256, 0, stream>>>(x, Mh, bias, out);
}
